// SelfAttention_71030169141392
// MI455X (gfx1250) — matrix operand order repaired and checked by reading, not compiled, not run on hardware
//
#include <hip/hip_runtime.h>
#include <hip/hip_bf16.h>

typedef __attribute__((ext_vector_type(16))) __bf16 v16bf;
typedef __attribute__((ext_vector_type(8)))  __bf16 v8bf;
typedef __attribute__((ext_vector_type(8)))  float  v8f;
typedef __attribute__((ext_vector_type(4)))  float  v4f;

#define EMB   1024
#define HEADS 16
#define HDIM  64
#define NB    8
#define SEQ   1024
#define MTOT  (NB * SEQ)   // 8192 rows in the flattened [token, emb] view

// ---------------------------------------------------------------------------
// WMMA fragment loaders (layouts per CDNA5 ISA §7.12.2, wave32)
// A 16x32 bf16: lane L(&15)=row M; lanes<16 hold K {0..7, 16..23},
//               lanes>=16 hold K {8..15, 24..31}; 8 contiguous bf16 per group.
// B 32x16 bf16: lane L(&15)=col N; lanes<16 hold K {0..7, 16..23},
//               lanes>=16 hold K {8..15, 24..31} (same order as A).
// C/D 16x16 f32: VGPR r -> row r + (lane>=16 ? 8 : 0); col = lane&15.
// ---------------------------------------------------------------------------
__device__ __forceinline__ v16bf load_a_frag(const __bf16* p, int rs) {
  const int lane = threadIdx.x & 31;
  const int row  = lane & 15;
  const int koff = (lane & 16) ? 8 : 0;
  const __bf16* b = p + row * rs;
  v8bf lo = *(const v8bf*)(b + koff);
  v8bf hi = *(const v8bf*)(b + 16 + koff);
  v16bf r;
#pragma unroll
  for (int i = 0; i < 8; ++i) { r[i] = lo[i]; r[8 + i] = hi[i]; }
  return r;
}

__device__ __forceinline__ v16bf load_b_frag(const __bf16* p, int rs) {
  const int lane  = threadIdx.x & 31;
  const int col   = lane & 15;
  const int kbase = (lane & 16) ? 8 : 0;
  const __bf16* b = p + col * rs + kbase;
  v8bf lo = *(const v8bf*)(b);
  v8bf hi = *(const v8bf*)(b + 16);
  v16bf r;
#pragma unroll
  for (int i = 0; i < 8; ++i) { r[i] = lo[i]; r[8 + i] = hi[i]; }
  return r;
}

__device__ __forceinline__ v8f wmma_bf16(v16bf a, v16bf b, v8f c) {
  return __builtin_amdgcn_wmma_f32_16x16x32_bf16(
      /*neg_a=*/false, a, /*neg_b=*/false, b,
      /*c_mod=*/(short)0, c, /*reuse_a=*/false, /*reuse_b=*/false);
}

// 16 f32 -> two v8bf (packed cvt), stored as two 16-B LDS writes
__device__ __forceinline__ void cvt_store16(__bf16* dst, const float* g) {
  v4f f0 = *(const v4f*)(g);
  v4f f1 = *(const v4f*)(g + 4);
  v4f f2 = *(const v4f*)(g + 8);
  v4f f3 = *(const v4f*)(g + 12);
  v8bf b0, b1;
#pragma unroll
  for (int i = 0; i < 4; ++i) {
    b0[i]     = (__bf16)f0[i];
    b0[4 + i] = (__bf16)f1[i];
    b1[i]     = (__bf16)f2[i];
    b1[4 + i] = (__bf16)f3[i];
  }
  *(v8bf*)(dst)     = b0;
  *(v8bf*)(dst + 8) = b1;
}

// ---------------------------------------------------------------------------
// GEMM: out[M,N] = A[M,K] @ W[N,K]^T  (nn.Linear semantics).
// W row-major is already K-major per output column == the "Bt" layout the
// B-fragment wants, so no transpose staging is needed.
// Block tile 128x128, BK=32, 8 waves; each wave computes 64x32 (4x2 WMMA
// tiles).  Double-buffered LDS: one barrier per k-step, staging overlaps WMMA.
// ---------------------------------------------------------------------------
template <bool A_BF16>
__device__ __forceinline__ void stage_tile(const void* __restrict__ Ap,
                                           const float* __restrict__ W,
                                           __bf16* __restrict__ sA,
                                           __bf16* __restrict__ sB,
                                           int mBase, int nBase, int k0, int tid) {
  const int srow = tid >> 1;        // 0..127
  const int scol = (tid & 1) * 16;  // 0 or 16
  if constexpr (A_BF16) {
    const __bf16* g = (const __bf16*)Ap + (size_t)(mBase + srow) * 1024 + k0 + scol;
    *(v8bf*)(sA + srow * 40 + scol)     = *(const v8bf*)g;
    *(v8bf*)(sA + srow * 40 + scol + 8) = *(const v8bf*)(g + 8);
    if (k0 + 64 < 1024) __builtin_prefetch(g + 64, 0, 1);
  } else {
    const float* g = (const float*)Ap + (size_t)(mBase + srow) * 1024 + k0 + scol;
    cvt_store16(sA + srow * 40 + scol, g);
    if (k0 + 64 < 1024) __builtin_prefetch(g + 64, 0, 1);
  }
  const float* gw = W + (size_t)(nBase + srow) * 1024 + k0 + scol;
  cvt_store16(sB + srow * 40 + scol, gw);
  if (k0 + 64 < 1024) __builtin_prefetch(gw + 64, 0, 1);
}

template <bool A_BF16, bool OUT_F32_BIAS>
__global__ __launch_bounds__(256) void gemm_wmma(const void* __restrict__ Ap,
                                                 const float* __restrict__ W,
                                                 const float* __restrict__ bias,
                                                 void* __restrict__ outp) {
  constexpr int Nd = 1024;
  __shared__ alignas(16) __bf16 sA[2][128 * 40];
  __shared__ alignas(16) __bf16 sB[2][128 * 40];

  const int tid  = threadIdx.x;
  const int lane = tid & 31;
  const int wave = tid >> 5;
  const int nBase = blockIdx.x * 128;
  const int mBase = blockIdx.y * 128;
  const int wm = wave >> 2;  // 0..1  (row half)
  const int wn = wave & 3;   // 0..3  (col quarter)

  const v8f vzero = {0.f, 0.f, 0.f, 0.f, 0.f, 0.f, 0.f, 0.f};
  v8f acc[4][2];
#pragma unroll
  for (int i = 0; i < 4; ++i)
#pragma unroll
    for (int j = 0; j < 2; ++j) acc[i][j] = vzero;

  stage_tile<A_BF16>(Ap, W, sA[0], sB[0], mBase, nBase, 0, tid);

  for (int kb = 0; kb < 32; ++kb) {
    __syncthreads();
    if (kb + 1 < 32)
      stage_tile<A_BF16>(Ap, W, sA[(kb + 1) & 1], sB[(kb + 1) & 1],
                         mBase, nBase, (kb + 1) * 32, tid);
    const __bf16* cA = sA[kb & 1];
    const __bf16* cB = sB[kb & 1];
    v16bf af[4];
#pragma unroll
    for (int i = 0; i < 4; ++i) af[i] = load_a_frag(cA + (wm * 64 + i * 16) * 40, 40);
    v16bf bfrag[2];
#pragma unroll
    for (int j = 0; j < 2; ++j) bfrag[j] = load_b_frag(cB + (wn * 32 + j * 16) * 40, 40);
#pragma unroll
    for (int i = 0; i < 4; ++i)
#pragma unroll
      for (int j = 0; j < 2; ++j) acc[i][j] = wmma_bf16(af[i], bfrag[j], acc[i][j]);
  }

  const int rb = (lane & 16) ? 8 : 0;
#pragma unroll
  for (int i = 0; i < 4; ++i)
#pragma unroll
    for (int j = 0; j < 2; ++j) {
      const int col  = nBase + wn * 32 + j * 16 + (lane & 15);
      const int row0 = mBase + wm * 64 + i * 16 + rb;
      if constexpr (OUT_F32_BIAS) {
        float* o = (float*)outp;
        const float bv = bias[col];
#pragma unroll
        for (int r = 0; r < 8; ++r)
          o[(size_t)(row0 + r) * Nd + col] = acc[i][j][r] + bv;
      } else {
        __bf16* o = (__bf16*)outp;
#pragma unroll
        for (int r = 0; r < 8; ++r)
          o[(size_t)(row0 + r) * Nd + col] = (__bf16)acc[i][j][r];
      }
    }
}

// ---------------------------------------------------------------------------
// Attention: one block per (batch n, head h, 32-query block).
// Scores 32x1024 live in LDS f32 (132 KB -- CDNA5's 320 KB LDS makes this
// possible).  QK^T and P@V via bf16 WMMA, softmax in f32 with wave32 shuffles.
// K/V chunk staging is double-buffered (one barrier per chunk).  Attention
// probabilities stream straight to d_out as b128 stores; context (bf16) to ws.
// ---------------------------------------------------------------------------
__global__ __launch_bounds__(256) void attn_wmma(const __bf16* __restrict__ Q,
                                                 const __bf16* __restrict__ Kt,
                                                 const __bf16* __restrict__ V,
                                                 const int* __restrict__ mask,
                                                 float* __restrict__ attnOut,
                                                 __bf16* __restrict__ ctx) {
  __shared__ alignas(16) float  sS[32 * 1032];     // scores, padded rows
  __shared__ alignas(16) __bf16 sQ[32 * 72];
  __shared__ alignas(16) __bf16 sKV[2][64 * 72];   // K chunks / V^T tiles

  const int tid = threadIdx.x, lane = tid & 31, wave = tid >> 5;
  const int qb = blockIdx.x, h = blockIdx.y, n = blockIdx.z;
  const int q0 = qb * 32;
  const size_t rowbase = (size_t)n * SEQ;  // row base into [8192 x 1024]
  const int hcol = h * HDIM;

  // stage Q block (32 x 64 bf16)
  {
    const int r = tid >> 3;         // 0..31
    const int c = (tid & 7) * 8;    // 0..56
    const __bf16* g = Q + (rowbase + q0 + r) * EMB + hcol + c;
    *(v8bf*)(sQ + r * 72 + c) = *(const v8bf*)g;
  }

  // K-chunk stager: 64 keys x 64 dims
  auto stageK = [&](int c, __bf16* buf) {
    const int r  = tid >> 2;          // 0..63 (key row in chunk)
    const int cc = (tid & 3) * 16;    // 0,16,32,48
    const __bf16* g = Kt + (rowbase + c * 64 + r) * EMB + hcol + cc;
    *(v8bf*)(buf + r * 72 + cc)     = *(const v8bf*)g;
    *(v8bf*)(buf + r * 72 + cc + 8) = *(const v8bf*)(g + 8);
  };
  // V^T-tile stager: 32 keys x 64 dims, stored transposed [d][key], pad 40
  auto stageV = [&](int ks, __bf16* buf) {
    const int key = tid >> 3;        // 0..31
    const int ds  = (tid & 7) * 8;   // 0..56
    const __bf16* g = V + (rowbase + ks * 32 + key) * EMB + hcol + ds;
    v8bf v = *(const v8bf*)g;
#pragma unroll
    for (int j = 0; j < 8; ++j) buf[(ds + j) * 40 + key] = v[j];
  };

  stageK(0, sKV[0]);
  __syncthreads();

  const int wm = wave >> 2;  // 0..1 : query sub-block
  const int wn = wave & 3;   // 0..3 : key/dim sub-block
  v16bf aq[2];
  aq[0] = load_a_frag(sQ + (wm * 16) * 72 + 0, 72);
  aq[1] = load_a_frag(sQ + (wm * 16) * 72 + 32, 72);

  const v8f vzero = {0.f, 0.f, 0.f, 0.f, 0.f, 0.f, 0.f, 0.f};
  const int rb = (lane & 16) ? 8 : 0;

  // ---- Phase 1: scores = QK^T / 8, masked, into LDS ----
  for (int c = 0; c < 16; ++c) {  // 16 chunks of 64 keys
    if (c + 1 < 16) stageK(c + 1, sKV[(c + 1) & 1]);
    const __bf16* cK = sKV[c & 1];
    v8f acc = vzero;
    acc = wmma_bf16(aq[0], load_b_frag(cK + (wn * 16) * 72 + 0, 72), acc);
    acc = wmma_bf16(aq[1], load_b_frag(cK + (wn * 16) * 72 + 32, 72), acc);
    const int kcol = c * 64 + wn * 16 + (lane & 15);
    const int mv = mask[n * SEQ + kcol];
#pragma unroll
    for (int r = 0; r < 8; ++r) {
      float e = acc[r] * 0.125f;        // 1/sqrt(64)
      if (mv == 0) e = -1e20f;
      sS[(wm * 16 + rb + r) * 1032 + kcol] = e;
    }
    __syncthreads();
  }

  // ---- Phase 2: softmax rows + stream attention to d_out (b128) ----
  float* aout = attnOut + (((size_t)(n * HEADS + h) * SEQ) + q0) * SEQ;
  for (int rr = 0; rr < 4; ++rr) {
    const int row = wave * 4 + rr;
    float* srow = sS + row * 1032;
    float m = -3.4e38f;
    for (int i0 = lane * 4; i0 < SEQ; i0 += 128) {
      v4f x = *(const v4f*)(srow + i0);
      m = fmaxf(m, fmaxf(fmaxf(x[0], x[1]), fmaxf(x[2], x[3])));
    }
#pragma unroll
    for (int off = 16; off > 0; off >>= 1) m = fmaxf(m, __shfl_xor(m, off, 32));
    float s = 0.f;
    for (int i0 = lane * 4; i0 < SEQ; i0 += 128) {
      v4f x = *(const v4f*)(srow + i0);
      v4f p;
#pragma unroll
      for (int j = 0; j < 4; ++j) { p[j] = __expf(x[j] - m); s += p[j]; }
      *(v4f*)(srow + i0) = p;
    }
#pragma unroll
    for (int off = 16; off > 0; off >>= 1) s += __shfl_xor(s, off, 32);
    const float inv = 1.f / s;
    float* orow = aout + (size_t)row * SEQ;
    for (int i0 = lane * 4; i0 < SEQ; i0 += 128) {
      v4f p = *(const v4f*)(srow + i0);
#pragma unroll
      for (int j = 0; j < 4; ++j) p[j] *= inv;
      *(v4f*)(srow + i0) = p;
      *(v4f*)(orow + i0) = p;
    }
  }
  __syncthreads();

  // ---- Phase 3: context = P @ V_h (32 x 64) ----
  stageV(0, sKV[0]);
  v8f acc = vzero;
  for (int ks = 0; ks < 32; ++ks) {  // 32-key steps
    __syncthreads();
    if (ks + 1 < 32) stageV(ks + 1, sKV[(ks + 1) & 1]);
    // A fragment from f32 probabilities in LDS (b128 reads + packed cvt)
    const int prow = wm * 16 + (lane & 15);
    const int koff = (lane & 16) ? 8 : 0;
    const float* pr = sS + prow * 1032 + ks * 32;
    v4f x0 = *(const v4f*)(pr + koff);
    v4f x1 = *(const v4f*)(pr + koff + 4);
    v4f x2 = *(const v4f*)(pr + 16 + koff);
    v4f x3 = *(const v4f*)(pr + 16 + koff + 4);
    v16bf pa;
#pragma unroll
    for (int j = 0; j < 4; ++j) {
      pa[j]      = (__bf16)x0[j];
      pa[4 + j]  = (__bf16)x1[j];
      pa[8 + j]  = (__bf16)x2[j];
      pa[12 + j] = (__bf16)x3[j];
    }
    acc = wmma_bf16(pa, load_b_frag(sKV[ks & 1] + (wn * 16) * 40, 40), acc);
  }
  {
    const int col  = hcol + wn * 16 + (lane & 15);
    const int row0 = q0 + wm * 16 + rb;
#pragma unroll
    for (int r = 0; r < 8; ++r)
      ctx[(rowbase + row0 + r) * EMB + col] = (__bf16)acc[r];
  }
}

// ---------------------------------------------------------------------------
extern "C" void kernel_launch(void* const* d_in, const int* in_sizes, int n_in,
                              void* d_out, int out_size, void* d_ws, size_t ws_size,
                              hipStream_t stream) {
  (void)in_sizes; (void)n_in; (void)out_size; (void)ws_size;
  const float* value = (const float*)d_in[0];
  const float* key_  = (const float*)d_in[1];
  const float* query = (const float*)d_in[2];
  const int*   mask  = (const int*)d_in[3];
  const float* Wv    = (const float*)d_in[4];
  const float* Wk    = (const float*)d_in[5];
  const float* Wq    = (const float*)d_in[6];
  const float* Wo    = (const float*)d_in[7];
  const float* bo    = (const float*)d_in[8];

  // workspace: bf16 Q | K | V | context, each 8192x1024 (16 MB) => 64 MB total
  __bf16* qb = (__bf16*)d_ws;
  __bf16* kb = qb + (size_t)MTOT * EMB;
  __bf16* vb = kb + (size_t)MTOT * EMB;
  __bf16* cb = vb + (size_t)MTOT * EMB;

  float* outp = (float*)d_out;                     // (8,1024,1024)
  float* attn = outp + (size_t)MTOT * EMB;         // (8,16,1024,1024)

  dim3 gg(EMB / 128, MTOT / 128);  // (8, 64)
  gemm_wmma<false, false><<<gg, 256, 0, stream>>>(query, Wq, nullptr, qb);
  gemm_wmma<false, false><<<gg, 256, 0, stream>>>(key_,  Wk, nullptr, kb);
  gemm_wmma<false, false><<<gg, 256, 0, stream>>>(value, Wv, nullptr, vb);
  attn_wmma<<<dim3(SEQ / 32, HEADS, NB), 256, 0, stream>>>(qb, kb, vb, mask, attn, cb);
  gemm_wmma<true, true><<<gg, 256, 0, stream>>>(cb, Wo, bo, outp);
}
